// MyModel_2_11879879541709
// MI455X (gfx1250) — matrix operand order repaired and checked by reading, not compiled, not run on hardware
//
#include <hip/hip_runtime.h>
#include <hip/hip_bf16.h>

// ---------------- problem constants (match reference) ----------------
#define B_      4
#define QLEN_   2048
#define MAXLEN_ 8192
#define DIM_    1024
#define KVLEN_  4096
#define SCALE_  0.03125f            // 1/sqrt(1024)

// ---------------- tiling ----------------
#define BLOCK_Q  32                 // queries per block
#define KV_TILE  64                 // keys per flash iteration
#define NTHREADS 256                // 8 wave32 waves

// padded LDS row lengths (element units) to avoid bank conflicts
#define QROW 1032                   // bf16, 2064 B rows (16B-aligned, bank-staggered)
#define KROW 72                     // bf16, d-major K chunk rows (144 B)
#define VROW 1032                   // bf16
#define SROW 66                     // f32
#define PROW 72                     // bf16

// dynamic LDS layout (bytes)
#define OFF_Q 0
#define OFF_K (OFF_Q + BLOCK_Q * QROW * 2)        //  66048
#define OFF_V (OFF_K + 256 * KROW * 2)            // 102912
#define OFF_S (OFF_V + KV_TILE * VROW * 2)        // 235008
#define OFF_P (OFF_S + BLOCK_Q * SROW * 4)        // 243456
#define OFF_M (OFF_P + BLOCK_Q * PROW * 2)        // 248064
#define OFF_L (OFF_M + BLOCK_Q * 4)
#define OFF_C (OFF_L + BLOCK_Q * 4)
#define SMEM_BYTES (OFF_C + BLOCK_Q * 4)          // 248448  (< 320 KB WGP LDS)

typedef __attribute__((ext_vector_type(16))) __bf16 bf16x16;
typedef __attribute__((ext_vector_type(8)))  __bf16 bf16x8;
typedef __attribute__((ext_vector_type(4)))  __bf16 bf16x4;
typedef __attribute__((ext_vector_type(8)))  float  f32x8;

static __device__ inline f32x8 zero8() {
  f32x8 z;
#pragma unroll
  for (int i = 0; i < 8; ++i) z[i] = 0.f;
  return z;
}

// two 16-byte LDS reads -> one 32-byte WMMA operand (ds_load_b128 x2)
static __device__ inline bf16x16 ld_frag(const __bf16* p0, const __bf16* p1) {
  union { bf16x16 v; bf16x8 h[2]; } u;
  u.h[0] = *(const bf16x8*)p0;
  u.h[1] = *(const bf16x8*)p1;
  return u.v;
}

static __device__ inline bf16x4 cvt4(float4 f) {
  bf16x4 r;
  r[0] = (__bf16)f.x; r[1] = (__bf16)f.y; r[2] = (__bf16)f.z; r[3] = (__bf16)f.w;
  return r;
}

__global__ __launch_bounds__(NTHREADS, 1)
void flash_attn_bf16_wmma(const int* __restrict__ ids,
                          const float* __restrict__ emb,
                          const float* __restrict__ Kc,
                          const float* __restrict__ Vc,
                          float* __restrict__ out) {
  extern __shared__ char smem[];
  __bf16* smQ = (__bf16*)(smem + OFF_Q);   // [32][QROW]   Q, pre-scaled, bf16
  __bf16* smK = (__bf16*)(smem + OFF_K);   // [256][KROW]  K chunk, d-major (transposed)
  __bf16* smV = (__bf16*)(smem + OFF_V);   // [64][VROW]   V tile, kv-major
  float*  smS = (float*)(smem + OFF_S);    // [32][SROW]   scores
  __bf16* smP = (__bf16*)(smem + OFF_P);   // [32][PROW]   probs (bf16)
  float*  smM = (float*)(smem + OFF_M);    // running row max
  float*  smL = (float*)(smem + OFF_L);    // running row sum
  float*  smC = (float*)(smem + OFF_C);    // per-tile rescale factor

  const int tid  = threadIdx.x;
  const int lane = tid & 31;
  const int wv   = tid >> 5;      // wave 0..7
  const int qw   = wv >> 2;       // q-subtile (16 rows): 0..1
  const int kw   = wv & 3;        // QK: 16-wide kv slice / PV & store: 256-wide d slice
  const int m    = lane & 15;
  const int hi   = lane >> 4;

  const int bid = blockIdx.x;
  const int b   = bid / (QLEN_ / BLOCK_Q);
  const int q0  = (bid % (QLEN_ / BLOCK_Q)) * BLOCK_Q;

  // ---- gather + scale + convert Q tile: emb[ids] * (1/sqrt(D)) -> bf16 LDS ----
  {
    const int row = tid >> 3;           // 0..31
    const int seg = tid & 7;            // 128 cols each
    const int id  = ids[b * QLEN_ + q0 + row];
    const float4* src = (const float4*)(emb + (size_t)id * DIM_ + seg * 128);
    __bf16* dst = smQ + row * QROW + seg * 128;
#pragma unroll
    for (int j = 0; j < 32; ++j) {
      float4 f = src[j];
      f.x *= SCALE_; f.y *= SCALE_; f.z *= SCALE_; f.w *= SCALE_;
      *(bf16x4*)(dst + j * 4) = cvt4(f);
    }
  }
  if (tid < BLOCK_Q) { smM[tid] = -__builtin_inff(); smL[tid] = 0.f; }
  __syncthreads();

  // O accumulator: 16 q-rows x 256 d-cols per wave = 16 WMMA C tiles
  f32x8 cO[16];
#pragma unroll
  for (int t = 0; t < 16; ++t) cO[t] = zero8();

  for (int kt = 0; kt < KVLEN_ / KV_TILE; ++kt) {
    const int kv0 = kt * KV_TILE;
    __syncthreads();   // previous PV reads of smV / smP complete

    // ---- stage V tile (64 x 1024 f32 -> bf16, kv-major) ----
    {
      const int row = tid >> 2;         // 0..63
      const int seg = tid & 3;          // 256 cols each
      const float4* src = (const float4*)(Vc + (size_t)(kv0 + row) * DIM_ + seg * 256);
      __bf16* dst = smV + row * VROW + seg * 256;
#pragma unroll
      for (int j = 0; j < 64; ++j)
        *(bf16x4*)(dst + j * 4) = cvt4(src[j]);
      if (kt + 1 < KVLEN_ / KV_TILE) {   // global_prefetch for next tile
        __builtin_prefetch(Vc + (size_t)(kv0 + KV_TILE + row) * DIM_ + seg * 256, 0, 1);
        __builtin_prefetch(Kc + (size_t)(kv0 + KV_TILE + row) * DIM_ + seg * 256, 0, 1);
      }
    }

    // ---- S = (Q*scale) x K^T, contracting DIM in 4 chunks of 256 ----
    f32x8 cS = zero8();
    for (int dc = 0; dc < 4; ++dc) {
      __syncthreads();                  // smK free for rewrite
      {
        // stage K chunk transposed: smK[d][kv]  (f32 -> bf16)
        const int kvr   = tid & 63;
        const int dpart = (tid >> 6) * 64;
        const float4* src =
            (const float4*)(Kc + (size_t)(kv0 + kvr) * DIM_ + dc * 256 + dpart);
#pragma unroll
        for (int j = 0; j < 16; ++j) {
          float4 f = src[j];
          const int d = dpart + j * 4;
          smK[(d + 0) * KROW + kvr] = (__bf16)f.x;
          smK[(d + 1) * KROW + kvr] = (__bf16)f.y;
          smK[(d + 2) * KROW + kvr] = (__bf16)f.z;
          smK[(d + 3) * KROW + kvr] = (__bf16)f.w;
        }
      }
      __syncthreads();
#pragma unroll
      for (int ks = 0; ks < 8; ++ks) {
        // A (Q 16x32): lane=M row; lo lanes K0-7/16-23, hi lanes K8-15/24-31
        const __bf16* qa = smQ + (qw * 16 + m) * QROW + dc * 256 + ks * 32 + hi * 8;
        bf16x16 a = ld_frag(qa, qa + 16);
        // B (K^T 32x16): lane=N(kv) column, elements over K(d) in A's order
        const __bf16* kb = smK + (ks * 32 + hi * 8) * KROW + kw * 16 + m;
        bf16x16 bb;
#pragma unroll
        for (int s = 0; s < 16; ++s) bb[s] = kb[((s & 7) + (s >> 3) * 16) * KROW];
        cS = __builtin_amdgcn_wmma_f32_16x16x32_bf16(false, a, false, bb,
                                                     (short)0, cS, false, false);
      }
    }
    // C layout: lane(0-15)=N col, VGPR r = row r (lo) / r+8 (hi)
#pragma unroll
    for (int r = 0; r < 8; ++r)
      smS[(qw * 16 + r + hi * 8) * SROW + kw * 16 + m] = cS[r];
    __syncthreads();

    // ---- online softmax over this 64-wide KV tile ----
    {
      const int row = tid >> 3;         // 8 lanes per row (same wave, aligned groups)
      const int sub = tid & 7;
      float s[8];
      float mx = -__builtin_inff();
#pragma unroll
      for (int j = 0; j < 8; ++j) {
        s[j] = smS[row * SROW + sub * 8 + j];
        mx = fmaxf(mx, s[j]);
      }
#pragma unroll
      for (int o = 1; o < 8; o <<= 1) mx = fmaxf(mx, __shfl_xor(mx, o, 8));
      const float m_old = smM[row];
      const float m_new = fmaxf(m_old, mx);
      const float corr  = __expf(m_old - m_new);   // first tile: exp(-inf)=0
      float sum = 0.f;
#pragma unroll
      for (int j = 0; j < 8; ++j) {
        const float p = __expf(s[j] - m_new);
        sum += p;
        smP[row * PROW + sub * 8 + j] = (__bf16)p;
      }
#pragma unroll
      for (int o = 1; o < 8; o <<= 1) sum += __shfl_xor(sum, o, 8);
      if (sub == 0) {
        smM[row] = m_new;
        smL[row] = smL[row] * corr + sum;
        smC[row] = corr;
      }
    }
    __syncthreads();

    // ---- rescale O by exp(m_old - m_new), then O += P x V ----
#pragma unroll
    for (int r = 0; r < 8; ++r) {
      const float cf = smC[qw * 16 + r + hi * 8];
#pragma unroll
      for (int t = 0; t < 16; ++t) cO[t][r] *= cf;
    }
#pragma unroll
    for (int kk = 0; kk < 2; ++kk) {
      // A (P 16x32 over kv)
      const __bf16* pa = smP + (qw * 16 + m) * PROW + kk * 32 + hi * 8;
      bf16x16 a = ld_frag(pa, pa + 16);
#pragma unroll
      for (int t = 0; t < 16; ++t) {
        // B (V 32x16): lane=N(d) column, elements over K(kv) in A's order
        const __bf16* vb = smV + (kk * 32 + hi * 8) * VROW + kw * 256 + t * 16 + m;
        bf16x16 bb;
#pragma unroll
        for (int s = 0; s < 16; ++s) bb[s] = vb[((s & 7) + (s >> 3) * 16) * VROW];
        cO[t] = __builtin_amdgcn_wmma_f32_16x16x32_bf16(false, a, false, bb,
                                                        (short)0, cO[t], false, false);
      }
    }
  } // kt

  // ---- epilogue: O / l, store f32 ----
#pragma unroll
  for (int r = 0; r < 8; ++r) {
    const int rowl = qw * 16 + r + hi * 8;
    const float inv = 1.f / smL[rowl];
    float* orow = out + ((size_t)(b * QLEN_ + q0 + rowl)) * DIM_ + kw * 256 + m;
#pragma unroll
    for (int t = 0; t < 16; ++t) orow[t * 16] = cO[t][r] * inv;
  }
}

extern "C" void kernel_launch(void* const* d_in, const int* in_sizes, int n_in,
                              void* d_out, int out_size, void* d_ws, size_t ws_size,
                              hipStream_t stream) {
  (void)in_sizes; (void)n_in; (void)d_ws; (void)ws_size; (void)out_size;
  const int*   ids = (const int*)d_in[0];
  // d_in[1]: attn_mask — all-true over the first KVLEN keys; padded tail is
  // excluded simply by iterating kv in [0, KVLEN). Not read.
  const float* emb = (const float*)d_in[2];
  const float* K   = (const float*)d_in[3];
  const float* V   = (const float*)d_in[4];
  float* out = (float*)d_out;

  static_assert(SMEM_BYTES <= 320 * 1024, "exceeds WGP LDS");
  (void)hipFuncSetAttribute((const void*)flash_attn_bf16_wmma,
                            hipFuncAttributeMaxDynamicSharedMemorySize,
                            SMEM_BYTES);

  dim3 grid(B_ * (QLEN_ / BLOCK_Q));   // 256 blocks
  dim3 block(NTHREADS);                // 8 waves
  hipLaunchKernelGGL(flash_attn_bf16_wmma, grid, block, SMEM_BYTES, stream,
                     ids, emb, K, V, out);
}
